// MultiScaleFeatureFusion_79242146611957
// MI455X (gfx1250) — matrix operand order repaired and checked by reading, not compiled, not run on hardware
//
#include <hip/hip_runtime.h>
#include <hip/hip_bf16.h>

// ---------------------------------------------------------------------------
// MI455X (gfx1250) multi-scale GAT fusion.
// Core matrix work on v_wmma_f32_16x16x32_bf16 (wave32 WMMA), software
// pipelined so loads for k+32 are in flight while WMMAs consume k.
// ---------------------------------------------------------------------------

typedef __bf16 bf16_t;
typedef __attribute__((ext_vector_type(16))) __bf16 v16bf;
typedef __attribute__((ext_vector_type(8)))  __bf16 v8bf;
typedef __attribute__((ext_vector_type(8)))  float   v8f;

constexpr int kN  = 4096;
constexpr int kD  = 256;
constexpr int kH  = 4;
constexpr int kDH = 64;
constexpr int kE  = 65536;
constexpr int kNW = kN / 32;   // 128 mask words per row
constexpr int k3D = 3 * kD;    // 768

#define DEVFN __device__ __forceinline__

// ---------------- WMMA helpers (ISA 7.12.2 wave32 layouts) -----------------

DEVFN v8f wmma_bf16(v16bf a, v16bf b, v8f c) {
  // (neg_a, A, neg_b, B, c_mod, C, reuse_a, reuse_b)
  return __builtin_amdgcn_wmma_f32_16x16x32_bf16(false, a, false, b, (short)0, c,
                                                 false, false);
}

// A operand: 16x32 bf16 tile from row-major [.. x lda], Abase at (m0, k0).
// lane<16: row=lane, K = {0..7} U {16..23}; lane>=16: row=lane-16, K={8..15} U {24..31}
DEVFN v16bf ld_a16x32(const bf16_t* __restrict__ Abase, int lda, int lane) {
  const int m  = lane & 15;
  const int kb = (lane < 16) ? 0 : 8;
  const bf16_t* p = Abase + (size_t)m * lda + kb;
  v8bf lo = *reinterpret_cast<const v8bf*>(p);
  v8bf hi = *reinterpret_cast<const v8bf*>(p + 16);
  v16bf r;
#pragma unroll
  for (int e = 0; e < 8; ++e) { r[e] = lo[e]; r[e + 8] = hi[e]; }
  return r;
}

// B operand: 32x16 bf16 tile; storage is B^T row-major [Ncols x K] (ld = ldb),
// Bbase at (n0, k0). lane holds column n = lane&15; lanes<16 K={0..7} U {16..23}, >=16 K={8..15} U {24..31}.
DEVFN v16bf ld_b32x16(const bf16_t* __restrict__ Bbase, int ldb, int lane) {
  const int n  = lane & 15;
  const int kb = (lane < 16) ? 0 : 8;
  const bf16_t* p = Bbase + (size_t)n * ldb + kb;
  v8bf lo = *reinterpret_cast<const v8bf*>(p);
  v8bf hi = *reinterpret_cast<const v8bf*>(p + 16);
  v16bf r;
#pragma unroll
  for (int e = 0; e < 8; ++e) { r[e] = lo[e]; r[e + 8] = hi[e]; }
  return r;
}

// ---------------- Generic bf16 WMMA GEMM: C = A * B^T ----------------------
// A: [M x K] bf16 row-major (lda). B: [Nc x K] bf16 row-major (ldb) == B^T.
// Optional f32 output C[M x Nc] (+bias), optional bf16 transposed output
// Ct[Nc x M] (ldct). One wave per 16x64 tile, double-buffered k-loop.
__global__ void gemm_wmma_bf16(const bf16_t* __restrict__ A, int lda,
                               const bf16_t* __restrict__ B, int ldb,
                               float* __restrict__ C, int ldc,
                               bf16_t* __restrict__ Ct, int ldct,
                               const float* __restrict__ bias,
                               int M, int K, int Nc) {
  const int lane  = threadIdx.x & 31;
  const int wave  = (blockIdx.x * blockDim.x + threadIdx.x) >> 5;
  const int tilesN = Nc >> 6;
  const int tiles  = (M >> 4) * tilesN;
  if (wave >= tiles) return;
  const int ib = wave / tilesN;
  const int cc = wave % tilesN;
  const int m0 = ib << 4;
  const int n0 = cc << 6;

  const bf16_t* Arow = A + (size_t)m0 * lda;
  const bf16_t* Brow[4];
#pragma unroll
  for (int t = 0; t < 4; ++t) Brow[t] = B + (size_t)(n0 + t * 16) * ldb;

  const v8f vz = {0.f, 0.f, 0.f, 0.f, 0.f, 0.f, 0.f, 0.f};
  v8f acc[4] = {vz, vz, vz, vz};

  // prologue: tiles for k0 = 0
  v16bf a_cur = ld_a16x32(Arow, lda, lane);
  v16bf b_cur[4];
#pragma unroll
  for (int t = 0; t < 4; ++t) b_cur[t] = ld_b32x16(Brow[t], ldb, lane);

  int k0 = 0;
  for (; k0 + 32 < K; k0 += 32) {
    const int kn = k0 + 32;
    if (kn + 32 < K) __builtin_prefetch(Arow + kn + 32, 0, 3);
    // issue next-slice loads first, then consume current slice
    v16bf a_nxt = ld_a16x32(Arow + kn, lda, lane);
    v16bf b_nxt[4];
#pragma unroll
    for (int t = 0; t < 4; ++t) b_nxt[t] = ld_b32x16(Brow[t] + kn, ldb, lane);
#pragma unroll
    for (int t = 0; t < 4; ++t) acc[t] = wmma_bf16(a_cur, b_cur[t], acc[t]);
    a_cur = a_nxt;
#pragma unroll
    for (int t = 0; t < 4; ++t) b_cur[t] = b_nxt[t];
  }
#pragma unroll
  for (int t = 0; t < 4; ++t) acc[t] = wmma_bf16(a_cur, b_cur[t], acc[t]);

  // C/D layout: VGPR r, lanes 0-15: M=r, N=lane; lanes 16-31: M=8+r, N=lane-16
  const int ncl   = lane & 15;
  const int mbase = m0 + ((lane >> 4) << 3);
#pragma unroll
  for (int t = 0; t < 4; ++t) {
    const int col = n0 + t * 16 + ncl;
    const float badd = bias ? bias[col] : 0.f;
    if (C) {
#pragma unroll
      for (int r = 0; r < 8; ++r)
        C[(size_t)(mbase + r) * ldc + col] = acc[t][r] + badd;
    }
    if (Ct) {
      v8bf o;
#pragma unroll
      for (int r = 0; r < 8; ++r) o[r] = (bf16_t)(acc[t][r] + badd);
      *reinterpret_cast<v8bf*>(Ct + (size_t)col * ldct + mbase) = o;  // contiguous in m
    }
  }
}

// ---------------- small utility kernels ------------------------------------

__global__ void f32_to_bf16_kernel(const float* __restrict__ src,
                                   bf16_t* __restrict__ dst, int n) {
  int i = blockIdx.x * blockDim.x + threadIdx.x;
  if (i < n) dst[i] = (bf16_t)src[i];
}

// edge (s,t): A[s,t]=1 ; M_local[t,s]=1 (mask[dst][src] = A^T)
__global__ void edge_scatter_kernel(const int* __restrict__ ei,
                                    unsigned* __restrict__ Abits,
                                    unsigned* __restrict__ Lbits) {
  int e = blockIdx.x * blockDim.x + threadIdx.x;
  if (e >= kE) return;
  int s = ei[e], t = ei[kE + e];
  atomicOr(&Abits[(size_t)s * kNW + (t >> 5)], 1u << (t & 31));
  atomicOr(&Lbits[(size_t)t * kNW + (s >> 5)], 1u << (s & 31));
}

__global__ void add_diag_kernel(unsigned* __restrict__ bits) {
  int i = blockIdx.x * blockDim.x + threadIdx.x;
  if (i < kN) bits[(size_t)i * kNW + (i >> 5)] |= 1u << (i & 31);
}

// A2 row s |= A row t for each edge (s,t)  (sparse boolean A@A)
__global__ void a2_expand_kernel(const int* __restrict__ ei,
                                 const unsigned* __restrict__ Abits,
                                 unsigned* __restrict__ A2bits) {
  int id = blockIdx.x * blockDim.x + threadIdx.x;
  if (id >= kE * kNW) return;
  int e = id >> 7;       // kNW == 128
  int w = id & (kNW - 1);
  int s = ei[e], t = ei[kE + e];
  unsigned word = Abits[(size_t)t * kNW + w];
  if (word) atomicOr(&A2bits[(size_t)s * kNW + w], word);
}

__global__ void or_words_kernel(const unsigned* __restrict__ a,
                                const unsigned* __restrict__ b,
                                unsigned* __restrict__ o, int n) {
  int i = blockIdx.x * blockDim.x + threadIdx.x;
  if (i < n) o[i] = a[i] | b[i];
}

// M_mid[i][j] = U[i][j] | U[j][i] | (i==j)
__global__ void build_mid_kernel(const unsigned* __restrict__ U,
                                 unsigned* __restrict__ Mid) {
  int id = blockIdx.x * blockDim.x + threadIdx.x;
  if (id >= kN * kNW) return;
  int i = id >> 7;
  int w = id & (kNW - 1);
  unsigned word = U[id];
  const int colw = i >> 5;
  const int colb = i & 31;
#pragma unroll 4
  for (int b = 0; b < 32; ++b) {
    int j = (w << 5) + b;
    word |= ((U[(size_t)j * kNW + colw] >> colb) & 1u) << b;
  }
  if ((i >> 5) == w) word |= 1u << (i & 31);
  Mid[id] = word;
}

// s_src[h][n] = sum_o zT[h][o][n]*a_src[h][o]  (zT bf16, (H, D, N))
__global__ void score_vec_kernel(const bf16_t* __restrict__ zT,
                                 const float* __restrict__ a_src,
                                 const float* __restrict__ a_dst,
                                 float* __restrict__ s_src,
                                 float* __restrict__ s_dst) {
  int id = blockIdx.x * blockDim.x + threadIdx.x;
  if (id >= kH * kN) return;
  int h = id >> 12;            // kN == 4096
  int n = id & (kN - 1);
  const bf16_t* zp = zT + (size_t)h * kD * kN + n;
  const float* as = a_src + h * kD;
  const float* ad = a_dst + h * kD;
  float ss = 0.f, sd = 0.f;
  for (int o = 0; o < kD; ++o) {
    float zv = (float)zp[(size_t)o * kN];
    ss += zv * as[o];
    sd += zv * ad[o];
  }
  s_src[id] = ss;
  s_dst[id] = sd;
}

// pass 1: masked row max + sum(exp) of leaky_relu(s_dst_i + s_src_j)
__global__ void gat_pass1_kernel(const unsigned* __restrict__ mask,  // null => all ones
                                 const float* __restrict__ s_src,
                                 const float* __restrict__ s_dst,
                                 float* __restrict__ mrow,
                                 float* __restrict__ lrow) {
  int id = blockIdx.x * blockDim.x + threadIdx.x;
  if (id >= kH * kN) return;
  int h = id >> 12;
  int i = id & (kN - 1);
  const float sd = s_dst[id];
  const float* ss = s_src + h * kN;
  float m = -1e30f, l = 0.f;
  for (int jw = 0; jw < kNW; ++jw) {
    unsigned mw = mask ? mask[(size_t)i * kNW + jw] : 0xFFFFFFFFu;
    if (!mw) continue;
#pragma unroll 4
    for (int b = 0; b < 32; ++b) {
      if ((mw >> b) & 1u) {
        float sc = sd + ss[(jw << 5) + b];
        sc = sc > 0.f ? sc : 0.2f * sc;
        if (sc > m) { l = l * __expf(m - sc) + 1.f; m = sc; }
        else        { l += __expf(sc - m); }
      }
    }
  }
  mrow[id] = m;
  lrow[id] = l;
}

// pass 2: out[i] = relu( mean_h( alpha_h @ z_h ) + bias ).
// The 1/H head-mean and 1/l softmax normalizer are folded into the alpha
// values, so all 4 heads accumulate into ONE WMMA accumulator set (no
// per-head accsum, fewer WMMA<->VALU hazards). B tiles are loaded before the
// trans-heavy alpha construction so exp() hides the load latency.
__global__ void gat_pass2_kernel(const unsigned* __restrict__ mask,  // null => dense
                                 const float* __restrict__ s_src,
                                 const float* __restrict__ s_dst,
                                 const float* __restrict__ mrow,
                                 const float* __restrict__ lrow,
                                 const bf16_t* __restrict__ zT,      // (H, D, N)
                                 const float* __restrict__ bias,
                                 bf16_t* __restrict__ concat_out,    // (N, 3D)
                                 int scale_off) {
  const int lane = threadIdx.x & 31;
  const int wave = (blockIdx.x * blockDim.x + threadIdx.x) >> 5;
  if (wave >= (kN / 16) * 4) return;
  const int ib = wave >> 2;
  const int cc = wave & 3;
  const int m0 = ib << 4;
  const int i  = m0 + (lane & 15);       // this lane's alpha row
  const int kbA = (lane < 16) ? 0 : 8;   // A-operand K base for this lane

  const v8f vz = {0.f, 0.f, 0.f, 0.f, 0.f, 0.f, 0.f, 0.f};
  v8f acc[4] = {vz, vz, vz, vz};

  for (int h = 0; h < kH; ++h) {
    const float sd   = s_dst[h * kN + i];
    const float m    = mrow[h * kN + i];
    const float wgt  = 0.25f / lrow[h * kN + i];  // softmax norm * head mean
    const float* ss  = s_src + h * kN;
    const bf16_t* zh = zT + (size_t)h * kD * kN + (size_t)(cc * 64) * kN;

    for (int jb = 0; jb < kN; jb += 32) {
      // 1) issue the 8 b128 loads for the four z B-tiles of this j-slice
      v16bf b0 = ld_b32x16(zh + 0 * 16 * (size_t)kN + jb, kN, lane);
      v16bf b1 = ld_b32x16(zh + 1 * 16 * (size_t)kN + jb, kN, lane);
      v16bf b2 = ld_b32x16(zh + 2 * 16 * (size_t)kN + jb, kN, lane);
      v16bf b3 = ld_b32x16(zh + 3 * 16 * (size_t)kN + jb, kN, lane);
      if (jb + 32 < kN) __builtin_prefetch(zh + jb + 32, 0, 3);
      // 2) build the 16x32 alpha tile (A operand) -- long VALU/trans sequence
      //    that overlaps the loads above
      const unsigned mw = mask ? mask[(size_t)i * kNW + (jb >> 5)] : 0xFFFFFFFFu;
      v16bf a;
#pragma unroll
      for (int e = 0; e < 16; ++e) {
        const int jc = jb + ((e & 8) ? 16 : 0) + kbA + (e & 7);
        float sc = sd + ss[jc];
        sc = sc > 0.f ? sc : 0.2f * sc;
        float al = ((mw >> (jc & 31)) & 1u) ? __expf(sc - m) * wgt : 0.f;
        a[e] = (bf16_t)al;
      }
      // 3) consume
      acc[0] = wmma_bf16(a, b0, acc[0]);
      acc[1] = wmma_bf16(a, b1, acc[1]);
      acc[2] = wmma_bf16(a, b2, acc[2]);
      acc[3] = wmma_bf16(a, b3, acc[3]);
    }
  }

  const int mbase = m0 + ((lane >> 4) << 3);
#pragma unroll
  for (int t = 0; t < 4; ++t) {
    const int col = cc * 64 + t * 16 + (lane & 15);
    const float bv = bias[col];
#pragma unroll
    for (int r = 0; r < 8; ++r) {
      float v = acc[t][r] + bv;
      v = v > 0.f ? v : 0.f;  // relu
      concat_out[(size_t)(mbase + r) * k3D + scale_off + col] = (bf16_t)v;
    }
  }
}

// per-node attention over the 3 scales (s == 3, tiny: scalar kernel)
__global__ void mha_combine_kernel(const float* __restrict__ q,
                                   const float* __restrict__ k,
                                   const float* __restrict__ v,
                                   float* __restrict__ att) {
  int id = blockIdx.x * blockDim.x + threadIdx.x;
  if (id >= kN * kH) return;
  int n = id >> 2;
  int h = id & 3;
  const float* qp = q + (size_t)n * kD + h * kDH;
  float s[3];
#pragma unroll
  for (int sc = 0; sc < 3; ++sc) {
    const float* kp = k + (size_t)sc * kN * kD + (size_t)n * kD + h * kDH;
    float d = 0.f;
    for (int t = 0; t < kDH; ++t) d += qp[t] * kp[t];
    s[sc] = d * 0.125f;  // 1/sqrt(64)
  }
  float mm = fmaxf(s[0], fmaxf(s[1], s[2]));
  float e0 = __expf(s[0] - mm), e1 = __expf(s[1] - mm), e2 = __expf(s[2] - mm);
  float inv = 1.f / (e0 + e1 + e2);
  e0 *= inv; e1 *= inv; e2 *= inv;
  const float* v0 = v + (size_t)n * kD + h * kDH;
  const float* v1 = v0 + (size_t)kN * kD;
  const float* v2 = v1 + (size_t)kN * kD;
  float* op = att + (size_t)n * kD + h * kDH;
  for (int t = 0; t < kDH; ++t) op[t] = e0 * v0[t] + e1 * v1[t] + e2 * v2[t];
}

// LayerNorm(h)*g+b -> relu -> + x_att ; one 256-thread block per row
__global__ void ln_fuse_kernel(const float* __restrict__ hbuf,
                               const float* __restrict__ xatt,
                               const float* __restrict__ g,
                               const float* __restrict__ b,
                               float* __restrict__ out) {
  const int row = blockIdx.x;
  const int t   = threadIdx.x;  // 0..255
  float v = hbuf[(size_t)row * kD + t];
  float s1 = v, s2 = v * v;
  for (int off = 16; off; off >>= 1) {
    s1 += __shfl_down(s1, off, 32);
    s2 += __shfl_down(s2, off, 32);
  }
  __shared__ float red[16];
  const int wid = t >> 5;
  if ((t & 31) == 0) { red[wid] = s1; red[8 + wid] = s2; }
  __syncthreads();
  float S1 = 0.f, S2 = 0.f;
#pragma unroll
  for (int w = 0; w < 8; ++w) { S1 += red[w]; S2 += red[8 + w]; }
  const float mu  = S1 * (1.0f / kD);
  const float var = S2 * (1.0f / kD) - mu * mu;
  const float rs  = rsqrtf(var + 1e-5f);
  float hv = (v - mu) * rs * g[t] + b[t];
  hv = hv > 0.f ? hv : 0.f;
  out[(size_t)row * kD + t] = hv + xatt[(size_t)row * kD + t];
}

// ---------------------------------------------------------------------------

static inline void launch_gemm(const bf16_t* A, int lda, const bf16_t* B, int ldb,
                               float* C, int ldc, bf16_t* Ct, int ldct,
                               const float* bias, int M, int K, int Nc,
                               hipStream_t s) {
  int tiles   = (M / 16) * (Nc / 64);
  int threads = tiles * 32;
  int grid    = (threads + 255) / 256;
  gemm_wmma_bf16<<<grid, 256, 0, s>>>(A, lda, B, ldb, C, ldc, Ct, ldct, bias, M, K, Nc);
}

extern "C" void kernel_launch(void* const* d_in, const int* in_sizes, int n_in,
                              void* d_out, int out_size, void* d_ws, size_t ws_size,
                              hipStream_t stream) {
  (void)in_sizes; (void)n_in; (void)out_size;
  const float* x  = (const float*)d_in[0];
  const int*   ei = (const int*)d_in[1];
  // GAT params: scale s -> W at 2+4s, a_src 3+4s, a_dst 4+4s, b 5+4s
  const float* Wq = (const float*)d_in[14]; const float* bq = (const float*)d_in[15];
  const float* Wk = (const float*)d_in[16]; const float* bk = (const float*)d_in[17];
  const float* Wv = (const float*)d_in[18]; const float* bv = (const float*)d_in[19];
  const float* Wo = (const float*)d_in[20]; const float* bo = (const float*)d_in[21];
  const float* Wf = (const float*)d_in[22]; const float* bf_ = (const float*)d_in[23];
  const float* lng = (const float*)d_in[24]; const float* lnb = (const float*)d_in[25];
  float* out = (float*)d_out;

  char* ws = (char*)d_ws;
  size_t off = 0;
  auto alloc = [&](size_t bytes) -> char* {
    char* p = ws + off;
    off += (bytes + 255) & ~(size_t)255;
    return p;
  };
  unsigned* mA   = (unsigned*)alloc((size_t)kN * kNW * 4);   // adjacency bits
  unsigned* mL   = (unsigned*)alloc((size_t)kN * kNW * 4);   // M_local bits
  unsigned* mA2  = (unsigned*)alloc((size_t)kN * kNW * 4);   // 2-hop bits
  unsigned* mU   = (unsigned*)alloc((size_t)kN * kNW * 4);   // A | A2
  unsigned* mMid = (unsigned*)alloc((size_t)kN * kNW * 4);   // M_mid bits
  bf16_t* x_bf   = (bf16_t*)alloc((size_t)kN * kD * 2);
  bf16_t* w_gat  = (bf16_t*)alloc((size_t)3 * kH * kD * kD * 2);
  bf16_t* w_qkvo = (bf16_t*)alloc((size_t)4 * kD * kD * 2);
  bf16_t* w_f    = (bf16_t*)alloc((size_t)kD * k3D * 2);
  bf16_t* zT     = (bf16_t*)alloc((size_t)kH * kD * kN * 2); // reused per scale
  float* ssrc    = (float*)alloc((size_t)kH * kN * 4);
  float* sdst    = (float*)alloc((size_t)kH * kN * 4);
  float* mrow    = (float*)alloc((size_t)kH * kN * 4);
  float* lrow    = (float*)alloc((size_t)kH * kN * 4);
  bf16_t* concat = (bf16_t*)alloc((size_t)kN * k3D * 2);     // [x_l | x_m | x_g] bf16
  float* qbuf    = (float*)alloc((size_t)kN * kD * 4);
  float* kbuf    = (float*)alloc((size_t)3 * kN * kD * 4);
  float* vbuf    = (float*)alloc((size_t)3 * kN * kD * 4);
  float* attb    = (float*)alloc((size_t)kN * kD * 4);
  bf16_t* att_bf = (bf16_t*)alloc((size_t)kN * kD * 2);
  float* xatt    = (float*)alloc((size_t)kN * kD * 4);
  float* hbuf    = (float*)alloc((size_t)kN * kD * 4);
  if (off > ws_size) return;  // insufficient scratch

  // --- masks ---------------------------------------------------------------
  hipMemsetAsync(mA, 0, (size_t)3 * ((size_t)kN * kNW * 4), stream);  // mA,mL,mA2 contiguous
  edge_scatter_kernel<<<kE / 256, 256, 0, stream>>>(ei, mA, mL);
  add_diag_kernel<<<kN / 256, 256, 0, stream>>>(mL);
  a2_expand_kernel<<<(kE * kNW) / 256, 256, 0, stream>>>(ei, mA, mA2);
  or_words_kernel<<<(kN * kNW) / 256, 256, 0, stream>>>(mA, mA2, mU, kN * kNW);
  build_mid_kernel<<<(kN * kNW) / 256, 256, 0, stream>>>(mU, mMid);

  // --- bf16 conversions ----------------------------------------------------
  auto cvt = [&](const float* s, bf16_t* d, int n) {
    f32_to_bf16_kernel<<<(n + 255) / 256, 256, 0, stream>>>(s, d, n);
  };
  cvt(x, x_bf, kN * kD);
  for (int s = 0; s < 3; ++s)
    cvt((const float*)d_in[2 + 4 * s], w_gat + (size_t)s * kH * kD * kD, kH * kD * kD);
  cvt(Wq, w_qkvo + 0 * (size_t)kD * kD, kD * kD);
  cvt(Wk, w_qkvo + 1 * (size_t)kD * kD, kD * kD);
  cvt(Wv, w_qkvo + 2 * (size_t)kD * kD, kD * kD);
  cvt(Wo, w_qkvo + 3 * (size_t)kD * kD, kD * kD);
  cvt(Wf, w_f, kD * k3D);

  // --- three GAT scales ----------------------------------------------------
  const unsigned* masks[3] = {mL, mMid, nullptr};
  for (int s = 0; s < 3; ++s) {
    const float* a_src = (const float*)d_in[3 + 4 * s];
    const float* a_dst = (const float*)d_in[4 + 4 * s];
    const float* bgat  = (const float*)d_in[5 + 4 * s];
    const bf16_t* wbf  = w_gat + (size_t)s * kH * kD * kD;
    // z_h = x @ W_h^T, stored transposed bf16 (H, D, N)
    for (int h = 0; h < kH; ++h)
      launch_gemm(x_bf, kD, wbf + (size_t)h * kD * kD, kD,
                  nullptr, 0, zT + (size_t)h * kD * kN, kN, nullptr,
                  kN, kD, kD, stream);
    score_vec_kernel<<<(kH * kN) / 256, 256, 0, stream>>>(zT, a_src, a_dst, ssrc, sdst);
    gat_pass1_kernel<<<(kH * kN) / 256, 256, 0, stream>>>(masks[s], ssrc, sdst, mrow, lrow);
    gat_pass2_kernel<<<((kN / 16) * 4 * 32) / 256, 256, 0, stream>>>(
        masks[s], ssrc, sdst, mrow, lrow, zT, bgat, concat, s * kD);
  }

  // --- cross-scale multihead attention -------------------------------------
  launch_gemm(x_bf, kD, w_qkvo + 0 * (size_t)kD * kD, kD,
              qbuf, kD, nullptr, 0, bq, kN, kD, kD, stream);
  for (int s = 0; s < 3; ++s) {
    const bf16_t* As = concat + (size_t)s * kD;  // lda = 3D slice
    launch_gemm(As, k3D, w_qkvo + 1 * (size_t)kD * kD, kD,
                kbuf + (size_t)s * kN * kD, kD, nullptr, 0, bk, kN, kD, kD, stream);
    launch_gemm(As, k3D, w_qkvo + 2 * (size_t)kD * kD, kD,
                vbuf + (size_t)s * kN * kD, kD, nullptr, 0, bv, kN, kD, kD, stream);
  }
  mha_combine_kernel<<<(kN * kH) / 256, 256, 0, stream>>>(qbuf, kbuf, vbuf, attb);
  cvt(attb, att_bf, kN * kD);
  launch_gemm(att_bf, kD, w_qkvo + 3 * (size_t)kD * kD, kD,
              xatt, kD, nullptr, 0, bo, kN, kD, kD, stream);

  // --- fusion MLP + LayerNorm + residual -----------------------------------
  launch_gemm(concat, k3D, w_f, k3D, hbuf, kD, nullptr, 0, bf_,
              kN, k3D, kD, stream);
  ln_fuse_kernel<<<kN, kD, 0, stream>>>(hbuf, xatt, lng, lnb, out);
}
